// LightGCN2_70128226009642
// MI455X (gfx1250) — matrix operand order repaired and checked by reading, not compiled, not run on hardware
//
#include <hip/hip_runtime.h>
#include <hip/hip_bf16.h>

#define NUM_USERS 50000
#define NUM_ITEMS 100000
#define N_NODES   150000
#define NNZ       2000000
#define LATENT    64
#define MINI      16
#define BATCH     1024

typedef __attribute__((ext_vector_type(16))) _Float16 v16h;
typedef __attribute__((ext_vector_type(8)))  _Float16 v8h;
typedef __attribute__((ext_vector_type(8)))  float    v8f;

// ---------------------------------------------------------------------------
// Init: x = acc = concat(emb_user, emb_item); xn = 0
// ---------------------------------------------------------------------------
__global__ void init64_k(const float* __restrict__ eu, const float* __restrict__ ei,
                         float* __restrict__ x, float* __restrict__ acc, float* __restrict__ xn) {
    int i = blockIdx.x * blockDim.x + threadIdx.x;
    if (i >= N_NODES * LATENT) return;
    int row = i >> 6, d = i & 63;
    float e = (row < NUM_USERS) ? eu[(size_t)row * LATENT + d]
                                : ei[(size_t)(row - NUM_USERS) * LATENT + d];
    x[i] = e; acc[i] = e; xn[i] = 0.0f;
}

__global__ void init16_k(const float* __restrict__ eu, const float* __restrict__ ei,
                         float* __restrict__ x, float* __restrict__ acc, float* __restrict__ xn) {
    int i = blockIdx.x * blockDim.x + threadIdx.x;
    if (i >= N_NODES * MINI) return;
    int row = i >> 4, d = i & 15;
    float e = (row < NUM_USERS) ? eu[(size_t)row * MINI + d]
                                : ei[(size_t)(row - NUM_USERS) * MINI + d];
    x[i] = e; acc[i] = e; xn[i] = 0.0f;
}

// ---------------------------------------------------------------------------
// Fused SpMM layer (latent 64 + mini 16):
//   lxn[rows[e]] += vals[e] * lx[cols[e]]   (all 32 lanes, 2 floats each)
//   mxn[rows[e]] += vals[e] * mx[cols[e]]   (lanes 0..15, 1 float each)
// One wave per edge; edge index is wave-uniform so rows/cols/vals become
// scalar (SMEM) loads. NNZ % 8 == 0, so no tail guard.
// ---------------------------------------------------------------------------
__global__ void spmm_fused_k(const int* __restrict__ rows, const int* __restrict__ cols,
                             const float* __restrict__ vals,
                             const float* __restrict__ lx, float* __restrict__ lxn,
                             const float* __restrict__ mx, float* __restrict__ mxn) {
    const int lane = threadIdx.x & 31;
    const int wave = threadIdx.x >> 5;
    const int edge = blockIdx.x * 8 + wave;          // wave-uniform

    // prefetch a future edge's gather row (global_prefetch_b8)
    int pe = edge + 4096;
    if (pe < NNZ) {
        int pc = cols[pe];                           // scalar load
        __builtin_prefetch(lx + (size_t)pc * LATENT + lane * 2, 0, 0);
    }

    const int   r = rows[edge];                      // scalar loads
    const int   c = cols[edge];
    const float v = vals[edge];

    const int d = lane * 2;
    float2 xv = *(const float2*)(lx + (size_t)c * LATENT + d);
    float* dst = lxn + (size_t)r * LATENT + d;
    unsafeAtomicAdd(dst,     v * xv.x);
    unsafeAtomicAdd(dst + 1, v * xv.y);

    if (lane < MINI) {
        float mv = mx[(size_t)c * MINI + lane];
        unsafeAtomicAdd(mxn + (size_t)r * MINI + lane, v * mv);
    }
}

// acc += xn; x = xn; xn = 0  (prepares next layer)
__global__ void accum_k(float* __restrict__ acc, float* __restrict__ x,
                        float* __restrict__ xn, int n) {
    int i = blockIdx.x * blockDim.x + threadIdx.x;
    if (i >= n) return;
    float v = xn[i];
    acc[i] += v; x[i] = v; xn[i] = 0.0f;
}

// ---------------------------------------------------------------------------
// Build f16 GEMM operands:  emb = (light_acc + upsample(mm_acc)) / 4
// upsample nearest: dim d takes mini dim d>>2
// ---------------------------------------------------------------------------
__global__ void build_items_k(const float* __restrict__ lacc, const float* __restrict__ macc,
                              _Float16* __restrict__ I16) {
    int i = blockIdx.x * blockDim.x + threadIdx.x;
    if (i >= NUM_ITEMS * LATENT) return;
    int item = i >> 6, d = i & 63;
    size_t row = (size_t)(NUM_USERS + item);
    float v = 0.25f * (lacc[row * LATENT + d] + macc[row * MINI + (d >> 2)]);
    I16[i] = (_Float16)v;
}

__global__ void build_users_k(const float* __restrict__ lacc, const float* __restrict__ macc,
                              const int* __restrict__ users, _Float16* __restrict__ U16) {
    int i = blockIdx.x * blockDim.x + threadIdx.x;
    if (i >= BATCH * LATENT) return;
    int b = i >> 6, d = i & 63;
    size_t u = (size_t)users[b];
    float v = 0.25f * (lacc[u * LATENT + d] + macc[u * MINI + (d >> 2)]);
    U16[i] = (_Float16)v;
}

// ---------------------------------------------------------------------------
// Rating GEMM: out[1024 x 100000] = sigmoid(U @ I^T + bu + bi)
// One wave computes a 16x16 tile via 2x v_wmma_f32_16x16x32_f16 (K=64),
// sweeping 25 N-tiles while keeping A fragments in registers.
// ---------------------------------------------------------------------------
#define NT_PER_WAVE 25
#define N_TILES     (NUM_ITEMS / 16)                 // 6250
#define N_CHUNKS    (N_TILES / NT_PER_WAVE)          // 250

__global__ void rating_gemm_k(const _Float16* __restrict__ U16,  // [1024][64]
                              const _Float16* __restrict__ I16,  // [100000][64]
                              const float* __restrict__ bias_user,
                              const float* __restrict__ bias_item,
                              const int* __restrict__ users,
                              float* __restrict__ out) {
    const int lane  = threadIdx.x & 31;
    const int wave  = threadIdx.x >> 5;
    const int chunk = blockIdx.x * 8 + wave;         // wave-uniform: EXEC stays full
    if (chunk >= N_CHUNKS) return;
    const int mbase = blockIdx.y * 16;
    const int n0    = chunk * NT_PER_WAVE * 16;

    const int mrow = lane & 15;   // A row (M) / B col (N) served by this lane
    const int hi   = lane >> 4;

    // --- A fragments (16x32 f16 layout, ISA 7.12.2), K = 0..31 and 32..63 ---
    const _Float16* arow = U16 + (size_t)(mbase + mrow) * LATENT;
    v16h a0, a1;
    {
        v8h l0 = *(const v8h*)(arow + hi * 8);
        v8h h0 = *(const v8h*)(arow + 16 + hi * 8);
        v8h l1 = *(const v8h*)(arow + 32 + hi * 8);
        v8h h1 = *(const v8h*)(arow + 48 + hi * 8);
        #pragma unroll
        for (int i = 0; i < 8; ++i) {
            a0[i] = l0[i]; a0[8 + i] = h0[i];
            a1[i] = l1[i]; a1[8 + i] = h1[i];
        }
    }

    // bias_user for the 8 output rows this lane owns (loop-invariant)
    float bu[8];
    #pragma unroll
    for (int r = 0; r < 8; ++r)
        bu[r] = bias_user[users[mbase + hi * 8 + r]];

    for (int t = 0; t < NT_PER_WAVE; ++t) {
        const int nb = n0 + t * 16;
        const _Float16* brow = I16 + (size_t)(nb + mrow) * LATENT;
        v16h b0, b1;
        {
            v8h x0 = *(const v8h*)(brow + hi * 8);
            v8h x1 = *(const v8h*)(brow + 16 + hi * 8);
            v8h y0 = *(const v8h*)(brow + 32 + hi * 8);
            v8h y1 = *(const v8h*)(brow + 48 + hi * 8);
            #pragma unroll
            for (int i = 0; i < 8; ++i) {
                b0[i] = x0[i]; b0[8 + i] = x1[i];
                b1[i] = y0[i]; b1[8 + i] = y1[i];
            }
        }

        v8f c = {};
        c = __builtin_amdgcn_wmma_f32_16x16x32_f16(false, a0, false, b0,
                                                   (short)0, c, false, false);
        c = __builtin_amdgcn_wmma_f32_16x16x32_f16(false, a1, false, b1,
                                                   (short)0, c, false, false);

        const int   ncol = nb + mrow;      // D layout: col = lane&15
        const float bi   = bias_item[ncol];
        #pragma unroll
        for (int r = 0; r < 8; ++r) {      // D layout: row = hi*8 + r
            int   mg = mbase + hi * 8 + r;
            float v  = c[r] + bu[r] + bi;
            // fast sigmoid: v_exp_f32 + v_rcp_f32 (avoid IEEE div chains)
            float e  = __expf(-v);
            out[(size_t)mg * NUM_ITEMS + ncol] = __builtin_amdgcn_rcpf(1.0f + e);
        }
    }
}

// ---------------------------------------------------------------------------
extern "C" void kernel_launch(void* const* d_in, const int* in_sizes, int n_in,
                              void* d_out, int out_size, void* d_ws, size_t ws_size,
                              hipStream_t stream) {
    const float* emb_user    = (const float*)d_in[0];
    const float* emb_item    = (const float*)d_in[1];
    const float* bias_user   = (const float*)d_in[2];
    const float* bias_item   = (const float*)d_in[3];
    const float* mm_emb_user = (const float*)d_in[4];
    const float* mm_emb_item = (const float*)d_in[5];
    const float* graph_vals  = (const float*)d_in[6];
    const int*   users       = (const int*)d_in[7];
    const int*   graph_rows  = (const int*)d_in[8];
    const int*   graph_cols  = (const int*)d_in[9];
    float* out = (float*)d_out;

    // Propagate scratch lives inside d_out (400 MB) until the final GEMM
    // overwrites it; f16 GEMM operands live in d_ws.
    float* lx   = out;                                   // 9.6 M f32
    float* lxn  = out + (size_t)N_NODES * LATENT;        // 9.6 M
    float* lacc = out + (size_t)2 * N_NODES * LATENT;    // 9.6 M
    float* mx   = out + (size_t)3 * N_NODES * LATENT;    // 2.4 M
    float* mxn  = mx  + (size_t)N_NODES * MINI;
    float* macc = mxn + (size_t)N_NODES * MINI;

    _Float16* I16 = (_Float16*)d_ws;                     // 6.4 M halfs
    _Float16* U16 = I16 + (size_t)NUM_ITEMS * LATENT;    // 65536 halfs

    const int B = 256;
    const int n64 = N_NODES * LATENT;                    // 9,600,000
    const int n16 = N_NODES * MINI;                      // 2,400,000

    init64_k<<<(n64 + B - 1) / B, B, 0, stream>>>(emb_user, emb_item, lx, lacc, lxn);
    init16_k<<<(n16 + B - 1) / B, B, 0, stream>>>(mm_emb_user, mm_emb_item, mx, macc, mxn);

    for (int layer = 0; layer < 3; ++layer) {
        spmm_fused_k<<<NNZ / 8, B, 0, stream>>>(graph_rows, graph_cols, graph_vals,
                                                lx, lxn, mx, mxn);
        accum_k<<<(n64 + B - 1) / B, B, 0, stream>>>(lacc, lx, lxn, n64);
        accum_k<<<(n16 + B - 1) / B, B, 0, stream>>>(macc, mx, mxn, n16);
    }

    build_items_k<<<(NUM_ITEMS * LATENT + B - 1) / B, B, 0, stream>>>(lacc, macc, I16);
    build_users_k<<<(BATCH * LATENT + B - 1) / B, B, 0, stream>>>(lacc, macc, users, U16);

    dim3 grid((N_CHUNKS + 7) / 8, BATCH / 16);           // (32, 64), 8 waves/block
    rating_gemm_k<<<grid, 256, 0, stream>>>(U16, I16, bias_user, bias_item, users, out);
}
